// HierarchicalClustering_25658134626778
// MI455X (gfx1250) — matrix operand order repaired and checked by reading, not compiled, not run on hardware
//
#include <hip/hip_runtime.h>
#include <hip/hip_bf16.h>
#include <math.h>

#define Bsz 4096
#define Dd  1024
#define Hh  1024
#define Ll  4
#define Nn  2
#define KC  64

typedef __attribute__((ext_vector_type(16))) __bf16 v16bf;
typedef __attribute__((ext_vector_type(8)))  __bf16 v8bf;
typedef __attribute__((ext_vector_type(8)))  float  v8f;
typedef __attribute__((ext_vector_type(4)))  unsigned int v4u;
typedef __attribute__((ext_vector_type(8)))  int v8i;
typedef __attribute__((ext_vector_type(4)))  int v4i;

#ifndef __has_builtin
#define __has_builtin(x) 0
#endif
#if __has_builtin(__builtin_amdgcn_tensor_load_to_lds) && \
    __has_builtin(__builtin_amdgcn_s_wait_tensorcnt)
#define USE_TDM 1
#else
#define USE_TDM 0
#endif

union FragU { v16bf v; v8bf h[2]; };

// ---- A fragment: 16x32 bf16, lane = l, row M = base+(l&15).
// ISA layout: VGPR v holds K = (v>>2)*16 + (l>>4)*8 + (v&3)*2 (+1)
// => per lane two contiguous 8-half chunks at k and k+16.
__device__ __forceinline__ v16bf fragA(const __bf16* lds, int rowbase, int kk, int lane) {
  int r  = rowbase + (lane & 15);
  int hh = (lane >> 4) * 8;
  const __bf16* p = lds + r * KC + kk + hh;
  FragU u;
  u.h[0] = *(const v8bf*)(p);
  u.h[1] = *(const v8bf*)(p + 16);
  return u.v;
}

// ---- B fragment: 32x16 bf16 stored transposed in LDS as [N][K].
// per lane two 8-half chunks at k=(l>>4)*8 and k+16 (same order as A).
__device__ __forceinline__ v16bf fragB(const __bf16* lds, int rowbase, int kk, int lane) {
  int r = rowbase + (lane & 15);
  const __bf16* p = lds + r * KC + kk + (lane >> 4) * 8;
  FragU u;
  u.h[0] = *(const v8bf*)(p);
  u.h[1] = *(const v8bf*)(p + 16);
  return u.v;
}

__device__ __forceinline__ v8f wmma_bf16(v16bf a, v16bf b, v8f c) {
  return __builtin_amdgcn_wmma_f32_16x16x32_bf16(false, a, false, b, (short)0, c,
                                                 false, false);
}

#if USE_TDM
// Issue one TDM 2D-tile load: tile is (rows x KC) bf16 elements, row stride
// ld_elems, packed contiguously into LDS at lds_byte_addr.
// D# group0: count=1 | lds_addr | global_addr(57b) | type=2.
// D# group1: data_size=2B | tensor_dim0=ld | tile_dim0=KC | tile_dim1=rows |
//            tensor_dim0_stride=ld.  Groups 2/3 (and extra group) null for 2D.
__device__ __forceinline__ void tdm_issue(unsigned lds_byte_addr,
                                          const __bf16* gptr, int ld_elems,
                                          int rows) {
  unsigned long long ga = (unsigned long long)(size_t)gptr;
  v4u g0;
  g0.x = 1u;                                     // count=1, user descriptor
  g0.y = lds_byte_addr;                          // LDS destination (bytes)
  g0.z = (unsigned)ga;                           // global_addr[31:0]
  g0.w = (unsigned)(ga >> 32) | 0x80000000u;     // global_addr[56:32] | type=2
  v8i g1;
  g1[0] = 0x00010000;                                        // data_size = 2 bytes
  g1[1] = (int)(((unsigned)ld_elems & 0xFFFFu) << 16);       // tensor_dim0 lo16
  g1[2] = (int)((((unsigned)ld_elems >> 16) & 0xFFFFu)       // tensor_dim0 hi16
                | 0xFFFF0000u);                              // tensor_dim1 = 65535
  g1[3] = (int)((unsigned)KC << 16);                         // tile_dim0 = KC
  g1[4] = rows;                                              // tile_dim1 = rows
  g1[5] = ld_elems;                                          // dim0_stride lo32
  g1[6] = 0;                                                 // dim0_stride hi, dim1_stride lo
  g1[7] = 0;
  v4i z4 = {0, 0, 0, 0};
  v8i z8 = {0, 0, 0, 0, 0, 0, 0, 0};
  __builtin_amdgcn_tensor_load_to_lds(g0, g1, z4, z4, z8, 0);
}
#endif

// 128(M) x 64(N) workgroup tile, 256 threads = 8 waves as 4x2.
// Each wave owns a 32x32 block (2x2 of 16x16 f32 accumulators).
// TDM double-buffered: wave0 issues next tile's DMA, waits TENSORcnt for the
// current one, all waves compute while the next tile streams into LDS.
__device__ __forceinline__ void gemm_tile(const __bf16* __restrict__ Ag, int lda,
                                          const __bf16* __restrict__ Bg, int ldb,
                                          int K, __bf16* ldsA, __bf16* ldsB,
                                          v8f (&acc)[2][2], int tid) {
  const int wid = tid >> 5, lane = tid & 31;
  const int wm = (wid >> 1) * 32, wn = (wid & 1) * 32;
#if USE_TDM
  const unsigned aB0 = (unsigned)(size_t)(void*)ldsA;
  const unsigned bB0 = (unsigned)(size_t)(void*)ldsB;
  const bool issuer = (tid < 32);
  __syncthreads();                       // LDS buffers free (prev call/use)
  if (issuer) {
    tdm_issue(aB0, Ag, lda, 128);
    tdm_issue(bB0, Bg, ldb, 64);
  }
  int buf = 0;
  for (int kc = 0; kc < K; kc += KC, buf ^= 1) {
    if (issuer) {
      if (kc + KC < K) {                 // prefetch next tile into other buffer
        unsigned nb = (unsigned)(buf ^ 1);
        tdm_issue(aB0 + nb * (128 * KC * 2), Ag + kc + KC, lda, 128);
        tdm_issue(bB0 + nb * (64 * KC * 2),  Bg + kc + KC, ldb, 64);
        __builtin_amdgcn_s_wait_tensorcnt(2);   // current pair complete
      } else {
        __builtin_amdgcn_s_wait_tensorcnt(0);
      }
    }
    __syncthreads();                     // current buffer visible to all waves
    const __bf16* lA = ldsA + buf * (128 * KC);
    const __bf16* lB = ldsB + buf * (64 * KC);
#pragma unroll
    for (int kk = 0; kk < KC; kk += 32) {
      v16bf a0 = fragA(lA, wm,      kk, lane);
      v16bf a1 = fragA(lA, wm + 16, kk, lane);
      v16bf b0 = fragB(lB, wn,      kk, lane);
      v16bf b1 = fragB(lB, wn + 16, kk, lane);
      acc[0][0] = wmma_bf16(a0, b0, acc[0][0]);
      acc[0][1] = wmma_bf16(a0, b1, acc[0][1]);
      acc[1][0] = wmma_bf16(a1, b0, acc[1][0]);
      acc[1][1] = wmma_bf16(a1, b1, acc[1][1]);
    }
    __syncthreads();                     // all waves done before buf is re-filled
  }
#else
  typedef __attribute__((ext_vector_type(4))) unsigned int u32x4;
  for (int kc = 0; kc < K; kc += KC) {
    __syncthreads();
    for (int c = tid; c < 128 * 8; c += 256) {
      int r = c >> 3, cc = c & 7;
      *(u32x4*)(ldsA + r * KC + cc * 8) =
          *(const u32x4*)(Ag + (size_t)r * lda + kc + cc * 8);
    }
    for (int c = tid; c < 64 * 8; c += 256) {
      int r = c >> 3, cc = c & 7;
      *(u32x4*)(ldsB + r * KC + cc * 8) =
          *(const u32x4*)(Bg + (size_t)r * ldb + kc + cc * 8);
    }
    if (kc + KC < K) {
      __builtin_prefetch(Ag + (size_t)(tid >> 1) * lda + kc + KC, 0, 1);
      __builtin_prefetch(Bg + (size_t)(tid & 63) * ldb + kc + KC, 0, 1);
    }
    __syncthreads();
#pragma unroll
    for (int kk = 0; kk < KC; kk += 32) {
      v16bf a0 = fragA(ldsA, wm,      kk, lane);
      v16bf a1 = fragA(ldsA, wm + 16, kk, lane);
      v16bf b0 = fragB(ldsB, wn,      kk, lane);
      v16bf b1 = fragB(ldsB, wn + 16, kk, lane);
      acc[0][0] = wmma_bf16(a0, b0, acc[0][0]);
      acc[0][1] = wmma_bf16(a0, b1, acc[0][1]);
      acc[1][0] = wmma_bf16(a1, b0, acc[1][0]);
      acc[1][1] = wmma_bf16(a1, b1, acc[1][1]);
    }
  }
#endif
}

__device__ __forceinline__ void zero_acc(v8f (&acc)[2][2]) {
#pragma unroll
  for (int i = 0; i < 2; i++)
#pragma unroll
    for (int j = 0; j < 2; j++)
#pragma unroll
      for (int r = 0; r < 8; r++) acc[i][j][r] = 0.0f;
}

// ---------------- spectral norm: one workgroup (1024 thr) per matrix ----------
__global__ void __launch_bounds__(1024)
k_sigma(const float* __restrict__ leaf_W, const float* __restrict__ noleaf_W,
        const float* __restrict__ root_W, float* __restrict__ inv_sigma) {
  __shared__ float su[1024], sv[1024], red[1024];
  const int m = blockIdx.x, t = threadIdx.x;
  const float* W = (m < 4) ? (leaf_W + (size_t)m * Hh * Dd)
                 : (m < 6) ? (noleaf_W + (size_t)(m - 4) * Hh * Hh)
                           : root_W;
  unsigned hsh = (unsigned)t * 2654435761u + 12345u;
  hsh ^= hsh >> 13; hsh *= 0x5bd1e995u; hsh ^= hsh >> 15;
  su[t] = (float)(hsh & 0xFFFF) / 32768.0f - 1.0f;
  __syncthreads();
  red[t] = su[t] * su[t]; __syncthreads();
  for (int s = 512; s > 0; s >>= 1) { if (t < s) red[t] += red[t + s]; __syncthreads(); }
  float nrm = sqrtf(red[0]) + 1e-12f;
  su[t] /= nrm; __syncthreads();

  float sigma = 1.0f;
  for (int it = 0; it < 5; it++) {
    float a = 0.0f;                       // v = W^T u (thread t walks column t)
    for (int r = 0; r < 1024; r++) a += W[(size_t)r * 1024 + t] * su[r];
    sv[t] = a; __syncthreads();
    red[t] = a * a; __syncthreads();
    for (int s = 512; s > 0; s >>= 1) { if (t < s) red[t] += red[t + s]; __syncthreads(); }
    nrm = sqrtf(red[0]) + 1e-12f;
    sv[t] /= nrm; __syncthreads();
    a = 0.0f;                             // u = W v
    for (int c = 0; c < 1024; c++) a += W[(size_t)t * 1024 + c] * sv[c];
    su[t] = a; __syncthreads();
    red[t] = a * a; __syncthreads();
    for (int s = 512; s > 0; s >>= 1) { if (t < s) red[t] += red[t + s]; __syncthreads(); }
    nrm = sqrtf(red[0]);
    sigma = nrm;                          // sigma = u.(Wv) = ||Wv||
    su[t] /= (nrm + 1e-12f); __syncthreads();
  }
  if (t == 0) inv_sigma[m] = 1.0f / sigma;
}

// ---------------- converts ---------------------------------------------------
__global__ void k_cvt(const float* __restrict__ src, __bf16* __restrict__ dst, size_t n) {
  size_t i0 = (size_t)blockIdx.x * blockDim.x + threadIdx.x;
  size_t stride = (size_t)gridDim.x * blockDim.x;
  for (size_t i = i0; i < n; i += stride) dst[i] = (__bf16)src[i];
}

__global__ void k_scale_cvt(const float* __restrict__ src, __bf16* __restrict__ dst,
                            size_t n, size_t per_mat, int sig_base,
                            const float* __restrict__ inv_sigma) {
  size_t i0 = (size_t)blockIdx.x * blockDim.x + threadIdx.x;
  size_t stride = (size_t)gridDim.x * blockDim.x;
  for (size_t i = i0; i < n; i += stride)
    dst[i] = (__bf16)(src[i] * inv_sigma[sig_base + (int)(i / per_mat)]);
}

// ---------------- assign: warp per batch row ---------------------------------
__global__ void __launch_bounds__(256)
k_assign(const float* __restrict__ x, const float* __restrict__ centers,
         float* __restrict__ assign_out, float* __restrict__ assign_ws) {
  int b = (int)((blockIdx.x * (size_t)blockDim.x + threadIdx.x) >> 5);
  int lane = threadIdx.x & 31;
  if (b >= Bsz) return;
  const float* xr = x + (size_t)b * Dd;
  float s[Ll] = {0.f, 0.f, 0.f, 0.f};
  for (int d = lane; d < Dd; d += 32) {
    float xv = xr[d];
#pragma unroll
    for (int l = 0; l < Ll; l++) { float df = xv - centers[l * Dd + d]; s[l] += df * df; }
  }
#pragma unroll
  for (int off = 16; off > 0; off >>= 1)
#pragma unroll
    for (int l = 0; l < Ll; l++) s[l] += __shfl_down(s[l], off, 32);
  if (lane == 0) {
    float e[Ll], tot = 0.f;
#pragma unroll
    for (int l = 0; l < Ll; l++) { e[l] = expf(-s[l] / 20.0f); tot += e[l]; }  // 2*SIGMA
    float inv = 1.0f / (tot + 1e-6f);
#pragma unroll
    for (int l = 0; l < Ll; l++) {
      float a = e[l] * inv;
      assign_out[l * Bsz + b] = a;
      assign_ws[l * Bsz + b]  = a;
    }
  }
}

// ---------------- leaf: leaf[l,b,h] = assign * tanh(X Wl^T + bl) -------------
__global__ void __launch_bounds__(256)
k_leaf(const __bf16* __restrict__ Xb, const __bf16* __restrict__ Wb,
       const float* __restrict__ leaf_b, const float* __restrict__ assign,
       __bf16* __restrict__ leaf_act) {
  __shared__ __bf16 ldsA[2 * 128 * KC];
  __shared__ __bf16 ldsB[2 * 64 * KC];
  const int l = blockIdx.z;
  const int bbase = blockIdx.y * 128, hbase = blockIdx.x * 64;
  const int tid = threadIdx.x, wid = tid >> 5, lane = tid & 31;
  const int wm = (wid >> 1) * 32, wn = (wid & 1) * 32;
  v8f acc[2][2];
  zero_acc(acc);
  gemm_tile(Xb + (size_t)bbase * Dd, Dd,
            Wb + ((size_t)l * Hh + hbase) * Dd, Dd, Dd, ldsA, ldsB, acc, tid);
#pragma unroll
  for (int i = 0; i < 2; i++)
#pragma unroll
    for (int j = 0; j < 2; j++) {
      int hg = hbase + wn + j * 16 + (lane & 15);
      float bb = leaf_b[l * Hh + hg];
#pragma unroll
      for (int r = 0; r < 8; r++) {
        int mg = bbase + wm + i * 16 + r + (lane >> 4) * 8;
        float av = assign[l * Bsz + mg];
        leaf_act[((size_t)l * Bsz + mg) * Hh + hg] =
            (__bf16)(av * tanhf(acc[i][j][r] + bb));
      }
    }
}

// ---------------- gates: softmax_n(leaf . gate_W[n] + gate_b[n]) -------------
__global__ void __launch_bounds__(256)
k_gates(const __bf16* __restrict__ leaf_act, const float* __restrict__ gate_W,
        const float* __restrict__ gate_b, float* __restrict__ gates_out,
        float* __restrict__ gates_ws) {
  int idx = (int)((blockIdx.x * (size_t)blockDim.x + threadIdx.x) >> 5);  // l*B+b
  int lane = threadIdx.x & 31;
  if (idx >= Ll * Bsz) return;
  int l = idx >> 12, b = idx & (Bsz - 1);
  const __bf16* row = leaf_act + (size_t)idx * Hh;
  float s0 = 0.f, s1 = 0.f;
  for (int h = lane; h < Hh; h += 32) {
    float lv = (float)row[h];
    s0 += lv * gate_W[h];
    s1 += lv * gate_W[Hh + h];
  }
#pragma unroll
  for (int off = 16; off > 0; off >>= 1) {
    s0 += __shfl_down(s0, off, 32);
    s1 += __shfl_down(s1, off, 32);
  }
  if (lane == 0) {
    float z0 = s0 + gate_b[0], z1 = s1 + gate_b[1];
    float mx = fmaxf(z0, z1);
    float e0 = expf(z0 - mx), e1 = expf(z1 - mx);
    float inv = 1.0f / (e0 + e1);
    float g0 = e0 * inv, g1 = e1 * inv;
    gates_out[((size_t)l * Nn + 0) * Bsz + b] = g0;
    gates_out[((size_t)l * Nn + 1) * Bsz + b] = g1;
    gates_ws [((size_t)l * Nn + 0) * Bsz + b] = g0;
    gates_ws [((size_t)l * Nn + 1) * Bsz + b] = g1;
  }
}

// ---------------- noleaf: sum_l gate[l,n,b]*tanh(leaf[l] Wn^T + bn) ----------
__global__ void __launch_bounds__(256)
k_noleaf(const __bf16* __restrict__ leaf_act, const __bf16* __restrict__ Wb,
         const float* __restrict__ noleaf_b, const float* __restrict__ gates,
         __bf16* __restrict__ noleaf_act) {
  __shared__ __bf16 ldsA[2 * 128 * KC];
  __shared__ __bf16 ldsB[2 * 64 * KC];
  const int n = blockIdx.z;
  const int bbase = blockIdx.y * 128, hbase = blockIdx.x * 64;
  const int tid = threadIdx.x, wid = tid >> 5, lane = tid & 31;
  const int wm = (wid >> 1) * 32, wn = (wid & 1) * 32;
  float outv[2][2][8];
#pragma unroll
  for (int i = 0; i < 2; i++)
#pragma unroll
    for (int j = 0; j < 2; j++)
#pragma unroll
      for (int r = 0; r < 8; r++) outv[i][j][r] = 0.0f;

  for (int l = 0; l < Ll; l++) {
    v8f acc[2][2];
    zero_acc(acc);
    gemm_tile(leaf_act + ((size_t)l * Bsz + bbase) * Hh, Hh,
              Wb + ((size_t)n * Hh + hbase) * Hh, Hh, Hh, ldsA, ldsB, acc, tid);
#pragma unroll
    for (int i = 0; i < 2; i++)
#pragma unroll
      for (int j = 0; j < 2; j++) {
        int hg = hbase + wn + j * 16 + (lane & 15);
        float bb = noleaf_b[n * Hh + hg];
#pragma unroll
        for (int r = 0; r < 8; r++) {
          int mg = bbase + wm + i * 16 + r + (lane >> 4) * 8;
          float g = gates[((size_t)l * Nn + n) * Bsz + mg];
          outv[i][j][r] += g * tanhf(acc[i][j][r] + bb);
        }
      }
  }
#pragma unroll
  for (int i = 0; i < 2; i++)
#pragma unroll
    for (int j = 0; j < 2; j++) {
      int hg = hbase + wn + j * 16 + (lane & 15);
#pragma unroll
      for (int r = 0; r < 8; r++) {
        int mg = bbase + wm + i * 16 + r + (lane >> 4) * 8;
        noleaf_act[((size_t)n * Bsz + mg) * Hh + hg] = (__bf16)outv[i][j][r];
      }
    }
}

// ---------------- root: sum_n tanh(noleaf[n] Wr^T + br) ----------------------
__global__ void __launch_bounds__(256)
k_root(const __bf16* __restrict__ noleaf_act, const __bf16* __restrict__ Wb,
       const float* __restrict__ root_b, float* __restrict__ out) {
  __shared__ __bf16 ldsA[2 * 128 * KC];
  __shared__ __bf16 ldsB[2 * 64 * KC];
  const int bbase = blockIdx.y * 128, hbase = blockIdx.x * 64;
  const int tid = threadIdx.x, wid = tid >> 5, lane = tid & 31;
  const int wm = (wid >> 1) * 32, wn = (wid & 1) * 32;
  float outv[2][2][8];
#pragma unroll
  for (int i = 0; i < 2; i++)
#pragma unroll
    for (int j = 0; j < 2; j++)
#pragma unroll
      for (int r = 0; r < 8; r++) outv[i][j][r] = 0.0f;

  for (int n = 0; n < Nn; n++) {
    v8f acc[2][2];
    zero_acc(acc);
    gemm_tile(noleaf_act + ((size_t)n * Bsz + bbase) * Hh, Hh,
              Wb + (size_t)hbase * Hh, Hh, Hh, ldsA, ldsB, acc, tid);
#pragma unroll
    for (int i = 0; i < 2; i++)
#pragma unroll
      for (int j = 0; j < 2; j++) {
        int hg = hbase + wn + j * 16 + (lane & 15);
        float bb = root_b[hg];
#pragma unroll
        for (int r = 0; r < 8; r++) outv[i][j][r] += tanhf(acc[i][j][r] + bb);
      }
  }
#pragma unroll
  for (int i = 0; i < 2; i++)
#pragma unroll
    for (int j = 0; j < 2; j++) {
      int hg = hbase + wn + j * 16 + (lane & 15);
#pragma unroll
      for (int r = 0; r < 8; r++) {
        int mg = bbase + wm + i * 16 + r + (lane >> 4) * 8;
        out[(size_t)mg * Hh + hg] = outv[i][j][r];
      }
    }
}

// ---------------- host launcher ----------------------------------------------
extern "C" void kernel_launch(void* const* d_in, const int* in_sizes, int n_in,
                              void* d_out, int out_size, void* d_ws, size_t ws_size,
                              hipStream_t stream) {
  (void)in_sizes; (void)n_in; (void)out_size; (void)ws_size;
  const float* x        = (const float*)d_in[0];
  const float* centers  = (const float*)d_in[1];
  const float* leaf_W   = (const float*)d_in[2];
  const float* leaf_b   = (const float*)d_in[3];
  const float* noleaf_W = (const float*)d_in[4];
  const float* noleaf_b = (const float*)d_in[5];
  const float* root_W   = (const float*)d_in[6];
  const float* root_b   = (const float*)d_in[7];
  const float* gate_W   = (const float*)d_in[8];
  const float* gate_b   = (const float*)d_in[9];

  char* ws = (char*)d_ws;
  const size_t OFF_SIGMA   = 0;
  const size_t OFF_XB      = 256;
  const size_t OFF_LEAFWB  = OFF_XB      + (size_t)2 * Bsz * Dd;
  const size_t OFF_NOLWB   = OFF_LEAFWB  + (size_t)2 * Ll * Hh * Dd;
  const size_t OFF_ROOTWB  = OFF_NOLWB   + (size_t)2 * Nn * Hh * Hh;
  const size_t OFF_LEAFACT = OFF_ROOTWB  + (size_t)2 * Hh * Hh;
  const size_t OFF_NOACT   = OFF_LEAFACT + (size_t)2 * Ll * Bsz * Hh;
  const size_t OFF_ASSIGN  = OFF_NOACT   + (size_t)2 * Nn * Bsz * Hh;
  const size_t OFF_GATES   = OFF_ASSIGN  + (size_t)4 * Ll * Bsz;

  float*  inv_sigma  = (float*)(ws + OFF_SIGMA);
  __bf16* Xb         = (__bf16*)(ws + OFF_XB);
  __bf16* leafWb     = (__bf16*)(ws + OFF_LEAFWB);
  __bf16* noleafWb   = (__bf16*)(ws + OFF_NOLWB);
  __bf16* rootWb     = (__bf16*)(ws + OFF_ROOTWB);
  __bf16* leaf_act   = (__bf16*)(ws + OFF_LEAFACT);
  __bf16* noleaf_act = (__bf16*)(ws + OFF_NOACT);
  float*  assign_ws  = (float*)(ws + OFF_ASSIGN);
  float*  gates_ws   = (float*)(ws + OFF_GATES);

  float* outF       = (float*)d_out;
  float* root_out   = outF;                                // [B,H]
  float* assign_out = outF + (size_t)Bsz * Hh;             // [L,B]
  float* gates_out  = assign_out + (size_t)Ll * Bsz;       // [L,N,B]

  k_sigma<<<7, 1024, 0, stream>>>(leaf_W, noleaf_W, root_W, inv_sigma);
  k_cvt<<<2048, 256, 0, stream>>>(x, Xb, (size_t)Bsz * Dd);
  k_scale_cvt<<<2048, 256, 0, stream>>>(leaf_W, leafWb, (size_t)Ll * Hh * Dd,
                                        (size_t)Hh * Dd, 0, inv_sigma);
  k_scale_cvt<<<1024, 256, 0, stream>>>(noleaf_W, noleafWb, (size_t)Nn * Hh * Hh,
                                        (size_t)Hh * Hh, 4, inv_sigma);
  k_scale_cvt<<<512, 256, 0, stream>>>(root_W, rootWb, (size_t)Hh * Hh,
                                       (size_t)Hh * Hh, 6, inv_sigma);
  k_assign<<<(Bsz * 32) / 256, 256, 0, stream>>>(x, centers, assign_out, assign_ws);
  k_leaf<<<dim3(Hh / 64, Bsz / 128, Ll), 256, 0, stream>>>(Xb, leafWb, leaf_b,
                                                           assign_ws, leaf_act);
  k_gates<<<(Ll * Bsz * 32) / 256, 256, 0, stream>>>(leaf_act, gate_W, gate_b,
                                                     gates_out, gates_ws);
  k_noleaf<<<dim3(Hh / 64, Bsz / 128, Nn), 256, 0, stream>>>(leaf_act, noleafWb,
                                                             noleaf_b, gates_ws,
                                                             noleaf_act);
  k_root<<<dim3(Hh / 64, Bsz / 128, 1), 256, 0, stream>>>(noleaf_act, rootWb,
                                                          root_b, root_out);
}
